// MICO_MIL_22308060135474
// MI455X (gfx1250) — compile-verified, matrix operand order checked by reading, not run on hardware
//
#include <hip/hip_runtime.h>

// ---------------------------------------------------------------------------
// MI455X (gfx1250) implementation: bf16 WMMA (f32 accum) for all large GEMMs,
// async global->LDS staging (ASYNCcnt) for the GEMM A-panels, pre-transposed
// weights so B-fragments load as contiguous b128 from L2-resident global.
// ---------------------------------------------------------------------------

typedef __bf16 bf16;
typedef __attribute__((ext_vector_type(16))) __bf16 v16bf;
typedef __attribute__((ext_vector_type(8)))  float  v8f;

#define N_PTS 50000
#define MPAD  50048   // padded row count (multiple of 64)
#define NTOT  50008   // N + 8 appended cluster rows
#define DIN   1024
#define EDIM  512
#define NCLS  2

// D = A(16x32) * B(32x16) + C, bf16 in / f32 accum
__device__ __forceinline__ v8f wmma_bf16(v16bf a, v16bf b, v8f c) {
  return __builtin_amdgcn_wmma_f32_16x16x32_bf16(
      /*neg_a=*/false, a, /*neg_b=*/false, b,
      /*c_mod=*/(short)0, c, /*reuse_a=*/false, /*reuse_b=*/false);
}

// Fragment load: lane L holds row (L&15); halves split at K offset 0/8 and 16/24.
// Element j (0..7)  -> K = half8 + j        (VGPR0..3)
// Element j (8..15) -> K = half8 + 16 + j-8 (VGPR4..7)
// => two 16-byte contiguous loads per lane.
__device__ __forceinline__ v16bf frag_ld(const bf16* p, int half8) {
  union { v16bf v; uint4 q[2]; } f;
  f.q[0] = *(const uint4*)(p + half8);
  f.q[1] = *(const uint4*)(p + half8 + 16);
  return f.v;
}

// ---------------------------------------------------------------------------
// GEMM (NT): out = act(A[MxK]bf16 @ WT[NxK]bf16^T + bias[N] (+ resid[MxN]))
// Requires N == 512, K % 64 == 0.
// Block: 8 waves, 32 rows x 512 cols. A K-panel (32x64) is staged into LDS
// with GLOBAL_LOAD_ASYNC_TO_LDS_B128 (ASYNCcnt). Each wave: 32x64 sub-tile,
// 8 WMMAs per 32-K step (16 per staged panel / barrier).
// ---------------------------------------------------------------------------
__global__ __launch_bounds__(256) void gemm_bf16_nt_kernel(
    const bf16* __restrict__ A, const bf16* __restrict__ WT,
    const float* __restrict__ bias, const float* resid,
    float* outF, bf16* outB, int M, int K, int N, int act)
{
  __shared__ __attribute__((aligned(16))) bf16 lA[32][64];   // [row][k] panel
  const int tid = threadIdx.x, lane = tid & 31, wv = tid >> 5;
  const int row0 = blockIdx.x * 32;
  const int colw = wv * 64;                 // wave's 64 output columns
  const int half8 = (lane & 16) ? 8 : 0;
  const int half = lane >> 4, cn = lane & 15, rl = lane & 15;

  v8f acc[2][4];
  { v8f z = {}; for (int i = 0; i < 2; ++i) for (int j = 0; j < 4; ++j) acc[i][j] = z; }

  // staging assignment: thread t copies 16B: row (t&31), k-chunk (t>>5)*8
  const int sr = tid & 31, sc = (tid >> 5) * 8;
  const unsigned lds_dst = (unsigned)(size_t)(&lA[sr][sc]);
  const bf16* gsrc_base = A + (size_t)(row0 + sr) * K + sc;

  for (int kp = 0; kp < K; kp += 64) {
    // ---- async stage A panel 32x64 (one b128 per lane) ----
    {
      unsigned long long gaddr = (unsigned long long)(gsrc_base + kp);
      asm volatile("global_load_async_to_lds_b128 %0, %1, off"
                   :: "v"(lds_dst), "v"(gaddr) : "memory");
      if (kp + 64 < K)
        __builtin_prefetch(gsrc_base + kp + 64, 0, 1);
      asm volatile("s_wait_asynccnt 0x0" ::: "memory");
    }
    __syncthreads();
    // ---- 2 k-steps of WMMA from the panel ----
#pragma unroll
    for (int ks = 0; ks < 2; ++ks) {
      const int kloc = ks * 32;
      v16bf fa0 = frag_ld(&lA[rl][kloc], half8);
      v16bf fa1 = frag_ld(&lA[16 + rl][kloc], half8);
#pragma unroll
      for (int j = 0; j < 4; ++j) {
        v16bf fb = frag_ld(WT + (size_t)(colw + j * 16 + rl) * K + kp + kloc, half8);
        acc[0][j] = wmma_bf16(fa0, fb, acc[0][j]);
        acc[1][j] = wmma_bf16(fa1, fb, acc[1][j]);
      }
    }
    __syncthreads();
  }

  // ---- epilogue: bias + optional residual + activation, f32/bf16 stores ----
#pragma unroll
  for (int i = 0; i < 2; ++i) {
#pragma unroll
    for (int r = 0; r < 8; ++r) {
      int row = row0 + i * 16 + r + 8 * half;
      if (row >= M) continue;
#pragma unroll
      for (int j = 0; j < 4; ++j) {
        int col = colw + j * 16 + cn;
        float v = acc[i][j][r] + bias[col];
        if (resid) v += resid[(size_t)row * N + col];
        if (act == 1)      v = fmaxf(v, 0.f);
        else if (act == 2) v = (v > 0.f) ? v : 0.01f * v;
        if (outF) outF[(size_t)row * N + col] = v;
        if (outB) outB[(size_t)row * N + col] = (bf16)v;
      }
    }
  }
}

// ---------------------------------------------------------------------------
// Fused cluster assignment: S = pe@ce^T (WMMA), sim = -sqrt(d2)/scale,
// softmax over clusters, pe += assign@ce (WMMA).  One 16-row tile per wave.
// ---------------------------------------------------------------------------
__global__ __launch_bounds__(256) void assign_update_kernel(
    const bf16* __restrict__ peB, float* peF,
    const bf16* __restrict__ ceB,   // [64][512] bf16 (nc valid rows)
    const bf16* __restrict__ ceT,   // [512][64] bf16 (cols >= nc zeroed)
    const float* __restrict__ ce2, const float* __restrict__ scale,
    int nc, int M)
{
  __shared__ __attribute__((aligned(16))) float sS[8][16][64];
  __shared__ __attribute__((aligned(16))) bf16  sA[8][16][64];
  __shared__ float sNorm[8][16][2];
  const int tid = threadIdx.x, lane = tid & 31, wv = tid >> 5;
  const int row0 = blockIdx.x * 128 + wv * 16;
  const int half8 = (lane & 16) ? 8 : 0;
  const int half = lane >> 4, cn = lane & 15, rl = lane & 15;
  const int ntc = nc >> 4;   // 4,2,1 cluster column tiles

  // --- S = pe @ ce^T ---
  v8f acc[4];
  { v8f z = {}; for (int t = 0; t < 4; ++t) acc[t] = z; }
  const bf16* aRow = peB + (size_t)(row0 + rl) * EDIM;
  for (int kb = 0; kb < EDIM; kb += 32) {
    v16bf fa = frag_ld(aRow + kb, half8);
    for (int nt = 0; nt < ntc; ++nt) {
      v16bf fb = frag_ld(ceB + (size_t)(nt * 16 + rl) * EDIM + kb, half8);
      acc[nt] = wmma_bf16(fa, fb, acc[nt]);
    }
  }
  for (int nt = 0; nt < ntc; ++nt)
#pragma unroll
    for (int r = 0; r < 8; ++r)
      sS[wv][r + 8 * half][nt * 16 + cn] = acc[nt][r];

  // --- row norms ||pe||^2 (2 lanes per row) ---
  {
    const float* prow = peF + (size_t)(row0 + rl) * EDIM + half * 256;
    float s2 = 0.f;
    for (int k = 0; k < 256; ++k) { float v = prow[k]; s2 += v * v; }
    sNorm[wv][rl][half] = s2;
  }
  __syncthreads();

  // --- softmax(-sqrt(d2)/scale) over clusters, write bf16 assignment ---
  if (lane < 16) {
    const float pn2 = sNorm[wv][rl][0] + sNorm[wv][rl][1];
    const float inv_sc = 1.f / scale[0];
    float m = -3.0e38f;
    for (int n = 0; n < nc; ++n) {
      float d2 = fmaxf(pn2 + ce2[n] - 2.f * sS[wv][rl][n], 0.f);
      float sim = -sqrtf(d2) * inv_sc;
      sS[wv][rl][n] = sim;
      m = fmaxf(m, sim);
    }
    float se = 0.f;
    for (int n = 0; n < nc; ++n) se += expf(sS[wv][rl][n] - m);
    float inv = 1.f / se;
    for (int n = 0; n < nc; ++n) sA[wv][rl][n] = (bf16)(expf(sS[wv][rl][n] - m) * inv);
    for (int n = nc; n < 64; ++n) sA[wv][rl][n] = (bf16)0.f;
  }
  __syncthreads();

  // --- pe += assign @ ce (K padded to >=32 with zeros) ---
  const int ncp = (nc < 32) ? 32 : nc;
  for (int nt = 0; nt < 32; ++nt) {
    v8f d = {};
    for (int kb = 0; kb < ncp; kb += 32) {
      v16bf fa = frag_ld(&sA[wv][rl][kb], half8);
      v16bf fb = frag_ld(ceT + (size_t)(nt * 16 + rl) * 64 + kb, half8);
      d = wmma_bf16(fa, fb, d);
    }
#pragma unroll
    for (int r = 0; r < 8; ++r) {
      int row = row0 + r + 8 * half;
      if (row < M) {
        size_t idx = (size_t)row * EDIM + nt * 16 + cn;
        peF[idx] = peF[idx] + d[r];
      }
    }
  }
}

// Row-wise layernorm (in-place f32, writes bf16 shadow).
__global__ __launch_bounds__(128) void layernorm_kernel(
    float* x, bf16* xb, const float* __restrict__ g, const float* __restrict__ b)
{
  __shared__ float r1[128], r2[128];
  const int row = blockIdx.x;
  float* xr = x + (size_t)row * EDIM;
  float s = 0.f, s2 = 0.f;
  for (int c = threadIdx.x; c < EDIM; c += 128) { float v = xr[c]; s += v; s2 += v * v; }
  r1[threadIdx.x] = s; r2[threadIdx.x] = s2;
  __syncthreads();
  for (int st = 64; st > 0; st >>= 1) {
    if (threadIdx.x < st) { r1[threadIdx.x] += r1[threadIdx.x + st]; r2[threadIdx.x] += r2[threadIdx.x + st]; }
    __syncthreads();
  }
  float mean = r1[0] * (1.f / EDIM);
  float var  = r2[0] * (1.f / EDIM) - mean * mean;
  float rstd = rsqrtf(var + 1e-5f);
  for (int c = threadIdx.x; c < EDIM; c += 128) {
    float v = (xr[c] - mean) * rstd * g[c] + b[c];
    xr[c] = v;
    xb[(size_t)row * EDIM + c] = (bf16)v;
  }
}

// ce prep: bf16 copy, transposed copy (zero-padded to 64 clusters), row norms.
__global__ void prep_ce_kernel(const float* __restrict__ ceF,
                               bf16* ceB, bf16* ceT, float* ce2, int nc)
{
  for (int i = threadIdx.x; i < 64 * EDIM; i += blockDim.x) {
    int k = i >> 9, e = i & 511;
    float v = (k < nc) ? ceF[(size_t)k * EDIM + e] : 0.f;
    ceB[(size_t)k * EDIM + e] = (bf16)v;
    ceT[(size_t)e * 64 + k]   = (bf16)v;
  }
  if (threadIdx.x < 64) {
    int k = threadIdx.x;
    float s = 0.f;
    if (k < nc) { const float* r = ceF + (size_t)k * EDIM; for (int e = 0; e < EDIM; ++e) s += r[e] * r[e]; }
    ce2[k] = s;
  }
}

// ClusterReducer: ce_out = (relu(ce^T @ W1 + b1) @ W2 + b2)^T  (tiny, VALU)
__global__ void cluster_reduce_kernel(const float* __restrict__ ceIn, int nc, int nc2,
                                      const float* __restrict__ W1, const float* __restrict__ b1,
                                      const float* __restrict__ W2, const float* __restrict__ b2,
                                      float* __restrict__ ceOut)
{
  int e = blockIdx.x * blockDim.x + threadIdx.x;
  if (e >= EDIM) return;
  float t[64], h[64];
  for (int c = 0; c < nc; ++c) t[c] = ceIn[(size_t)c * EDIM + e];
  for (int j = 0; j < nc; ++j) {
    float a = b1[j];
    for (int c = 0; c < nc; ++c) a += t[c] * W1[c * nc + j];
    h[j] = fmaxf(a, 0.f);
  }
  for (int j2 = 0; j2 < nc2; ++j2) {
    float a = b2[j2];
    for (int j = 0; j < nc; ++j) a += h[j] * W2[j * nc2 + j2];
    ceOut[(size_t)j2 * EDIM + e] = a;
  }
}

// Append final 8 ce rows to pe buffers (emb = [pe; ce]).
__global__ void append_ce_kernel(const float* __restrict__ ceF, float* peF, bf16* peB)
{
  for (int i = threadIdx.x; i < 8 * EDIM; i += blockDim.x) {
    int r = i >> 9, c = i & 511;
    float v = ceF[(size_t)r * EDIM + c];
    size_t idx = (size_t)(N_PTS + r) * EDIM + c;
    peF[idx] = v; peB[idx] = (bf16)v;
  }
}

// Attention scores: tanh(proc@W1+b1)*sigmoid(proc@W2+b2) dot Was + bas.
// W1T/W2T pre-transposed [n][k] bf16 so B-fragments are K-contiguous.
__global__ __launch_bounds__(256) void attention_kernel(
    const bf16* __restrict__ prB,
    const bf16* __restrict__ W1T, const float* __restrict__ b1,
    const bf16* __restrict__ W2T, const float* __restrict__ b2,
    const float* __restrict__ Was, const float* __restrict__ bas,
    float* __restrict__ scores, int M)
{
  __shared__ float red[8][16][16];
  const int tid = threadIdx.x, lane = tid & 31, wv = tid >> 5;
  const int row0 = (blockIdx.x * 8 + wv) * 16;
  const int half8 = (lane & 16) ? 8 : 0;
  const int half = lane >> 4, cn = lane & 15, rl = lane & 15;
  float ps[8] = {0.f, 0.f, 0.f, 0.f, 0.f, 0.f, 0.f, 0.f};
  const bf16* aRow = prB + (size_t)(row0 + rl) * EDIM;

  for (int nt = 0; nt < 32; ++nt) {
    v8f at = {}, ag = {};
    const bf16* w1r = W1T + (size_t)(nt * 16 + rl) * EDIM;
    const bf16* w2r = W2T + (size_t)(nt * 16 + rl) * EDIM;
    for (int kb = 0; kb < EDIM; kb += 32) {
      v16bf fa = frag_ld(aRow + kb, half8);
      at = wmma_bf16(fa, frag_ld(w1r + kb, half8), at);
      ag = wmma_bf16(fa, frag_ld(w2r + kb, half8), ag);
    }
#pragma unroll
    for (int r = 0; r < 8; ++r) {
      int col = nt * 16 + cn;
      float tv = tanhf(at[r] + b1[col]);
      float gv = 1.f / (1.f + expf(-(ag[r] + b2[col])));
      ps[r] += tv * gv * Was[col];
    }
  }
#pragma unroll
  for (int r = 0; r < 8; ++r) red[wv][r + 8 * half][cn] = ps[r];
  __syncthreads();
  if (lane < 16) {
    float s = 0.f;
    for (int j = 0; j < 16; ++j) s += red[wv][lane][j];
    int row = row0 + lane;
    if (row < M) scores[row] = s + bas[0];
  }
}

__global__ void score_max_part_kernel(const float* __restrict__ scores, float* part, int M)
{
  __shared__ float r[256];
  float m = -3.0e38f;
  for (int i = blockIdx.x * blockDim.x + threadIdx.x; i < M; i += blockDim.x * gridDim.x)
    m = fmaxf(m, scores[i]);
  r[threadIdx.x] = m;
  __syncthreads();
  for (int st = 128; st > 0; st >>= 1) {
    if (threadIdx.x < st) r[threadIdx.x] = fmaxf(r[threadIdx.x], r[threadIdx.x + st]);
    __syncthreads();
  }
  if (threadIdx.x == 0) part[blockIdx.x] = r[0];
}

__global__ void score_max_fin_kernel(const float* __restrict__ part, float* gmax,
                                     float* slideAcc, float* sumE)
{
  if (threadIdx.x == 0) {
    float m = part[0];
    for (int j = 1; j < 64; ++j) m = fmaxf(m, part[j]);
    gmax[0] = m;
    sumE[0] = 0.f;
  }
  for (int c = threadIdx.x; c < EDIM; c += blockDim.x) slideAcc[c] = 0.f;
}

// softmax-weighted pooling: slideAcc += sum_r e_r * proc_r ; sumE += sum e_r
__global__ __launch_bounds__(256) void attn_accum_kernel(
    const float* __restrict__ scores, const float* __restrict__ prF,
    const float* __restrict__ gmax, float* slideAcc, float* sumE, int M)
{
  __shared__ float ew[32];
  const int r0 = blockIdx.x * 32;
  if (threadIdx.x < 32) {
    int row = r0 + threadIdx.x;
    ew[threadIdx.x] = (row < M) ? expf(scores[row] - gmax[0]) : 0.f;
  }
  __syncthreads();
  if (threadIdx.x == 0) {
    float s = 0.f;
    for (int j = 0; j < 32; ++j) s += ew[j];
    atomicAdd(sumE, s);
  }
  for (int c = threadIdx.x; c < EDIM; c += 256) {
    float a = 0.f;
    for (int j = 0; j < 32; ++j) {
      float w = ew[j];
      if (w != 0.f) a += w * prF[(size_t)(r0 + j) * EDIM + c];
    }
    atomicAdd(&slideAcc[c], a);
  }
}

__global__ __launch_bounds__(256) void final_head_kernel(
    const float* __restrict__ slideAcc, const float* __restrict__ sumE,
    const float* __restrict__ Wfin, const float* __restrict__ bfin,
    const float* __restrict__ Wcls, const float* __restrict__ bcls,
    float* __restrict__ out)
{
  __shared__ float sl[EDIM], fn[EDIM];
  float inv = 1.f / sumE[0];
  for (int c = threadIdx.x; c < EDIM; c += 256) sl[c] = slideAcc[c] * inv;
  __syncthreads();
  for (int o = threadIdx.x; o < EDIM; o += 256) {
    float a = bfin[o];
    for (int k = 0; k < EDIM; ++k) a += sl[k] * Wfin[(size_t)k * EDIM + o];
    fn[o] = fmaxf(a, 0.f);
  }
  __syncthreads();
  if (threadIdx.x < NCLS) {
    float a = bcls[threadIdx.x];
    for (int k = 0; k < EDIM; ++k) a += fn[k] * Wcls[k * NCLS + threadIdx.x];
    out[threadIdx.x] = a;
  }
}

__global__ void cvt_kernel(const float* __restrict__ in, bf16* __restrict__ out, long long n)
{
  for (long long i = blockIdx.x * (long long)blockDim.x + threadIdx.x; i < n;
       i += (long long)blockDim.x * gridDim.x)
    out[i] = (bf16)in[i];
}

// out[c][r] = in[r][c]  (in: [R][Cc] f32 -> out: [Cc][R] bf16)
__global__ void cvtT_kernel(const float* __restrict__ in, bf16* __restrict__ out, int R, int Cc)
{
  for (long long i = blockIdx.x * (long long)blockDim.x + threadIdx.x; i < (long long)R * Cc;
       i += (long long)blockDim.x * gridDim.x) {
    int r = (int)(i / Cc), c = (int)(i % Cc);
    out[(size_t)c * R + r] = (bf16)in[(size_t)r * Cc + c];
  }
}

// ---------------------------------------------------------------------------
// Host launcher
// ---------------------------------------------------------------------------
extern "C" void kernel_launch(void* const* d_in, const int* in_sizes, int n_in,
                              void* d_out, int out_size, void* d_ws, size_t ws_size,
                              hipStream_t stream) {
  (void)in_sizes; (void)n_in; (void)out_size; (void)ws_size;
  // Input leaves in setup_inputs() dict order (lists flattened in order).
  const float* x     = (const float*)d_in[0];
  const float* cc    = (const float*)d_in[1];
  const float* scale = (const float*)d_in[2];
  const float* Wpp   = (const float*)d_in[3];
  const float* bpp   = (const float*)d_in[4];
  const float* Wcp   = (const float*)d_in[5];
  const float* bcp   = (const float*)d_in[6];
  const float* fc1W  = (const float*)d_in[7];
  const float* fc1b  = (const float*)d_in[8];
  const float* fc2W  = (const float*)d_in[9];
  const float* fc2b  = (const float*)d_in[10];
  const float* lng   = (const float*)d_in[11];
  const float* lnb   = (const float*)d_in[12];
  const float* r1W[3] = {(const float*)d_in[13], (const float*)d_in[14], (const float*)d_in[15]};
  const float* r1b[3] = {(const float*)d_in[16], (const float*)d_in[17], (const float*)d_in[18]};
  const float* r2W[3] = {(const float*)d_in[19], (const float*)d_in[20], (const float*)d_in[21]};
  const float* r2b[3] = {(const float*)d_in[22], (const float*)d_in[23], (const float*)d_in[24]};
  const float* Wfp  = (const float*)d_in[25];
  const float* bfp  = (const float*)d_in[26];
  const float* agg  = (const float*)d_in[27];
  const float* agb  = (const float*)d_in[28];
  const float* Wat1 = (const float*)d_in[29];
  const float* bat1 = (const float*)d_in[30];
  const float* Wat2 = (const float*)d_in[31];
  const float* bat2 = (const float*)d_in[32];
  const float* Was  = (const float*)d_in[33];
  const float* bas  = (const float*)d_in[34];
  const float* Wfin = (const float*)d_in[35];
  const float* bfinp= (const float*)d_in[36];
  const float* Wcls = (const float*)d_in[37];
  const float* bcls = (const float*)d_in[38];

  // Workspace layout (256B aligned slices).
  char* base = (char*)d_ws;
  size_t off = 0;
  auto take = [&](size_t bytes) -> char* {
    char* p = base + off;
    off = (off + bytes + 255) & ~(size_t)255;
    return p;
  };
  bf16*  xb     = (bf16*)take((size_t)MPAD * DIN  * sizeof(bf16));   // also aliased as prF
  float* peF    = (float*)take((size_t)MPAD * EDIM * sizeof(float));
  bf16*  peB    = (bf16*)take((size_t)MPAD * EDIM * sizeof(bf16));
  bf16*  hB     = (bf16*)take((size_t)MPAD * EDIM * sizeof(bf16));   // also aliased as prB
  // all weights pre-transposed to [N][K] bf16 (B-fragments K-contiguous)
  bf16*  WppT   = (bf16*)take((size_t)EDIM * DIN * sizeof(bf16));
  bf16*  WcpT   = (bf16*)take((size_t)EDIM * DIN * sizeof(bf16));
  bf16*  W1T    = (bf16*)take((size_t)3 * EDIM * EDIM * sizeof(bf16));
  bf16*  W2T    = (bf16*)take((size_t)3 * EDIM * EDIM * sizeof(bf16));
  bf16*  WfpT   = (bf16*)take((size_t)EDIM * EDIM * sizeof(bf16));
  bf16*  Wat1T  = (bf16*)take((size_t)EDIM * EDIM * sizeof(bf16));
  bf16*  Wat2T  = (bf16*)take((size_t)EDIM * EDIM * sizeof(bf16));
  bf16*  ccB    = (bf16*)take((size_t)64 * DIN * sizeof(bf16));
  float* ceFa   = (float*)take((size_t)64 * EDIM * sizeof(float));
  float* ceFb   = (float*)take((size_t)64 * EDIM * sizeof(float));
  bf16*  ceB    = (bf16*)take((size_t)64 * EDIM * sizeof(bf16));
  bf16*  ceT    = (bf16*)take((size_t)EDIM * 64 * sizeof(bf16));
  float* ce2    = (float*)take(64 * sizeof(float));
  float* scores = (float*)take((size_t)MPAD * sizeof(float));
  float* partMx = (float*)take(128 * sizeof(float));
  float* slideA = (float*)take(EDIM * sizeof(float));
  float* sumE   = (float*)take(16 * sizeof(float));
  float* gmax   = partMx + 64;
  // aliases (dead buffers reused for final stage)
  float* prF = (float*)xb;   // x not needed after first GEMM
  bf16*  prB = hB;           // h not needed after last enhancement

  const int NCS_[4] = {64, 32, 16, 8};
  const int GX_PTS  = (N_PTS + 31) / 32;   // covers 50016 rows (< MPAD)

  // --- precision conversion / weight pre-transpose ---
  cvt_kernel<<<2048, 256, 0, stream>>>(x, xb, (long long)N_PTS * DIN);
  cvt_kernel<<<64,   256, 0, stream>>>(cc, ccB, (long long)64 * DIN);
  cvtT_kernel<<<512, 256, 0, stream>>>(Wpp, WppT, DIN, EDIM);
  cvtT_kernel<<<512, 256, 0, stream>>>(Wcp, WcpT, DIN, EDIM);
  for (int i = 0; i < 3; ++i) {
    cvtT_kernel<<<256, 256, 0, stream>>>(fc1W + (size_t)i * EDIM * EDIM,
                                         W1T + (size_t)i * EDIM * EDIM, EDIM, EDIM);
    cvtT_kernel<<<256, 256, 0, stream>>>(fc2W + (size_t)i * EDIM * EDIM,
                                         W2T + (size_t)i * EDIM * EDIM, EDIM, EDIM);
  }
  cvtT_kernel<<<256, 256, 0, stream>>>(Wfp, WfpT, EDIM, EDIM);
  cvtT_kernel<<<256, 256, 0, stream>>>(Wat1, Wat1T, EDIM, EDIM);
  cvtT_kernel<<<256, 256, 0, stream>>>(Wat2, Wat2T, EDIM, EDIM);

  // --- pe = leaky_relu(x @ Wpp + bpp), ce = leaky_relu(cc @ Wcp + bcp) ---
  gemm_bf16_nt_kernel<<<GX_PTS, 256, 0, stream>>>(xb, WppT, bpp, nullptr, peF, peB,
                                                  N_PTS, DIN, EDIM, 2);
  gemm_bf16_nt_kernel<<<2, 256, 0, stream>>>(ccB, WcpT, bcp, nullptr,
                                             ceFa, nullptr, 64, DIN, EDIM, 2);
  prep_ce_kernel<<<1, 256, 0, stream>>>(ceFa, ceB, ceT, ce2, 64);

  float* ceCur = ceFa;
  float* ceNxt = ceFb;
  for (int i = 0; i < 3; ++i) {
    const int nc = NCS_[i], nc2 = NCS_[i + 1];
    assign_update_kernel<<<(N_PTS + 127) / 128, 256, 0, stream>>>(
        peB, peF, ceB, ceT, ce2, scale, nc, N_PTS);
    layernorm_kernel<<<N_PTS, 128, 0, stream>>>(peF, peB, lng + i * EDIM, lnb + i * EDIM);
    gemm_bf16_nt_kernel<<<GX_PTS, 256, 0, stream>>>(
        peB, W1T + (size_t)i * EDIM * EDIM, fc1b + i * EDIM, nullptr,
        nullptr, hB, N_PTS, EDIM, EDIM, 1);
    gemm_bf16_nt_kernel<<<GX_PTS, 256, 0, stream>>>(
        hB, W2T + (size_t)i * EDIM * EDIM, fc2b + i * EDIM, peF,
        peF, peB, N_PTS, EDIM, EDIM, 0);
    cluster_reduce_kernel<<<2, 256, 0, stream>>>(ceCur, nc, nc2,
                                                 r1W[i], r1b[i], r2W[i], r2b[i], ceNxt);
    float* t = ceCur; ceCur = ceNxt; ceNxt = t;
    prep_ce_kernel<<<1, 256, 0, stream>>>(ceCur, ceB, ceT, ce2, nc2);
  }

  // --- emb = [pe; ce]; proc = layernorm(relu(emb @ Wfp + bfp)) ---
  append_ce_kernel<<<1, 256, 0, stream>>>(ceCur, peF, peB);
  gemm_bf16_nt_kernel<<<(NTOT + 31) / 32, 256, 0, stream>>>(
      peB, WfpT, bfp, nullptr, prF, nullptr, NTOT, EDIM, EDIM, 1);
  layernorm_kernel<<<NTOT, 128, 0, stream>>>(prF, prB, agg, agb);

  // --- gated attention pooling + head ---
  const int tiles = (NTOT + 15) / 16;
  attention_kernel<<<(tiles + 7) / 8, 256, 0, stream>>>(prB, Wat1T, bat1, Wat2T, bat2,
                                                        Was, bas, scores, NTOT);
  score_max_part_kernel<<<64, 256, 0, stream>>>(scores, partMx, NTOT);
  score_max_fin_kernel<<<1, 256, 0, stream>>>(partMx, gmax, slideA, sumE);
  attn_accum_kernel<<<(NTOT + 31) / 32, 256, 0, stream>>>(scores, prF, gmax, slideA, sumE, NTOT);
  final_head_kernel<<<1, 256, 0, stream>>>(slideA, sumE, Wfin, bfinp, Wcls, bcls,
                                           (float*)d_out);
}
